// MPNN_5643587027232
// MI455X (gfx1250) — compile-verified, matrix operand order checked by reading, not run on hardware
//
#include <hip/hip_runtime.h>

typedef __attribute__((ext_vector_type(16))) _Float16 v16h;
typedef __attribute__((ext_vector_type(8)))  float    v8f;

#define BATCH 16
#define NNODE 4096
#define KNBR  32
#define NTOT  (BATCH * NNODE)   // 65536 nodes

#define BR_BLOCKS 2048          // branch kernel grid
#define NODES_PER_WAVE (NTOT / (BR_BLOCKS * 8))   // = 4

// ---------------------------------------------------------------------------
// Kernel 1: h = relu(x @ We + be)   [65536,4] @ [4,12] -> fp16 [65536,16]
// (slots 12..15 zeroed so they act as K-padding for the WMMA A operand)
// ---------------------------------------------------------------------------
__global__ __launch_bounds__(256) void embed_kernel(
    const float* __restrict__ x, const float* __restrict__ We,
    const float* __restrict__ be, _Float16* __restrict__ h16)
{
    const int g = blockIdx.x * 256 + threadIdx.x;   // node id 0..65535
    const float x0 = x[g * 4 + 0];
    const float x1 = x[g * 4 + 1];
    const float x2 = x[g * 4 + 2];
    const float x3 = x[g * 4 + 3];
    v16h o;
#pragma unroll
    for (int j = 0; j < 12; ++j) {
        float a = be[j] + x0 * We[0 * 12 + j] + x1 * We[1 * 12 + j]
                        + x2 * We[2 * 12 + j] + x3 * We[3 * 12 + j];
        o[j] = (_Float16)fmaxf(a, 0.0f);
    }
    o[12] = o[13] = o[14] = o[15] = (_Float16)0.0f;
    *(v16h*)(h16 + (size_t)g * 16) = o;
}

// ---------------------------------------------------------------------------
// Kernel 2 (per branch): message + update.
// One wave handles one node at a time: 32 edges = 2 WMMA tiles of 16 edges.
// A tile (16x32 f16): K[0..11]=gathered hj, K[16..27]=rbf, rest zero.
// B tile (32x16 f16): Wm rows 0..11 at K rows 0..11, rows 12..23 at 16..27.
// Tiles are double-buffered in LDS so both tiles' loads/stages overlap.
// ---------------------------------------------------------------------------
__global__ __launch_bounds__(256) void branch_kernel(
    const float*    __restrict__ d,    // [NTOT, K, 2] (idx, dist)
    const _Float16* __restrict__ h16,  // [NTOT, 16]
    const float*    __restrict__ Wm,   // [24, 12]
    const float*    __restrict__ bm,   // [12]
    const float*    __restrict__ Wu,   // [24, 12]
    const float*    __restrict__ bu,   // [12]
    float*          __restrict__ out)  // [NTOT, 12]
{
    __shared__ _Float16 Wm_lds[32][16];
    __shared__ float    bm_lds[16];
    __shared__ float    Wu_lds[24 * 12];
    __shared__ float    bu_lds[12];
    __shared__ __align__(64) _Float16 hd[8][2][16][32];  // per-wave, per-tile
    __shared__ float    m_buf[8][16];

    const int tid = threadIdx.x;

    // Stage padded weights into LDS
    for (int i = tid; i < 512; i += 256) {
        const int r = i >> 4, c = i & 15;
        float v = 0.0f;
        if (c < 12) {
            if (r < 12)                 v = Wm[r * 12 + c];
            else if (r >= 16 && r < 28) v = Wm[(r - 4) * 12 + c];
        }
        Wm_lds[r][c] = (_Float16)v;
    }
    if (tid < 16) bm_lds[tid] = (tid < 12) ? bm[tid] : 0.0f;
    for (int i = tid; i < 288; i += 256) Wu_lds[i] = Wu[i];
    if (tid < 12) bu_lds[tid] = bu[tid];
    __syncthreads();

    const int  lane   = tid & 31;
    const int  w      = tid >> 5;
    const int  col    = lane & 15;
    const bool hiHalf = (lane >= 16);

    // Persistent B fragment: lane L holds column L&15; low half lanes hold
    // K rows {0..7,16..23}, high half lanes hold K rows {8..15,24..31}.
    v16h bfrag;
    {
        const int rb = hiHalf ? 8 : 0;
#pragma unroll
        for (int i = 0; i < 8; ++i) bfrag[i]     = Wm_lds[rb + i][col];
#pragma unroll
        for (int i = 0; i < 8; ++i) bfrag[8 + i] = Wm_lds[16 + rb + i][col];
    }

    const float step   = 0.3f / 11.0f;                    // RBF center spacing
    const float inv2s2 = 1.0f / (2.0f * 0.015f * 0.015f); // 1/(2*sigma^2)

    const int waveGlobal = blockIdx.x * 8 + w;
    const int g0 = waveGlobal * NODES_PER_WAVE;

    for (int n = 0; n < NODES_PER_WAVE; ++n) {
        const int g = g0 + n;
        const int bbase = g & ~(NNODE - 1);               // b * 4096

        // Prefetch next node's edge block (256 B = 2 cachelines) into caches
        if (n + 1 < NODES_PER_WAVE) {
            const char* nd = (const char*)(d + (size_t)(g + 1) * KNBR * 2);
            __builtin_prefetch(nd, 0, 3);
            __builtin_prefetch(nd + 128, 0, 3);
        }

        float m_acc = 0.0f;

#pragma unroll
        for (int t = 0; t < 2; ++t) {
            const int e = col;                            // edge within tile
            const int k = t * 16 + e;
            const float2 dv = ((const float2*)d)[(size_t)g * KNBR + k];

            if (!hiHalf) {
                // gather neighbor features -> A tile K[0..15] (12 + zero pad)
                const int idx = (int)dv.x;
                v16h hj = *(const v16h*)(h16 + (size_t)(bbase + idx) * 16);
                *(v16h*)(&hd[w][t][e][0]) = hj;
            } else {
                // RBF features -> A tile K[16..31] (12 + zero pad)
                const float dist = dv.y;
                v16h r;
#pragma unroll
                for (int j = 0; j < 12; ++j) {
                    const float df = dist - step * (float)j;
                    r[j] = (_Float16)__expf(-df * df * inv2s2);
                }
                r[12] = r[13] = r[14] = r[15] = (_Float16)0.0f;
                *(v16h*)(&hd[w][t][e][16]) = r;
            }
            // Same-wave LDS ops execute in order: stores above land before
            // the fragment loads below (DScnt tracked by compiler).

            // A fragment: lane L -> row L&15; K subset per half.
            v16h a;
            const int off = hiHalf ? 8 : 0;
#pragma unroll
            for (int i = 0; i < 8; ++i) a[i]     = hd[w][t][col][off + i];
#pragma unroll
            for (int i = 0; i < 8; ++i) a[8 + i] = hd[w][t][col][16 + off + i];

            v8f c = {};
            c = __builtin_amdgcn_wmma_f32_16x16x32_f16(
                    false, a, false, bfrag, (short)0, c, false, false);

            // bias + relu per edge, reduce over the tile's 16 edges
            const float bias = bm_lds[col];
            float s = 0.0f;
#pragma unroll
            for (int r8 = 0; r8 < 8; ++r8) s += fmaxf(c[r8] + bias, 0.0f);
            // add the other half (rows 8..15 / 0..7) via SWAPX16 swizzle
            const int sw = __builtin_amdgcn_ds_swizzle(__float_as_int(s), 0x401F);
            s += __int_as_float(sw);
            m_acc += s;
        }

        if (!hiHalf) m_buf[w][col] = m_acc;               // m[col] for node g

        // update: sigmoid([h, m] @ Wu + bu), 12 lanes -> 12 outputs
        if (lane < 12) {
            const _Float16* hg = h16 + (size_t)g * 16;
            float acc = bu_lds[lane];
#pragma unroll
            for (int ci = 0; ci < 12; ++ci)
                acc += (float)hg[ci] * Wu_lds[ci * 12 + lane];
#pragma unroll
            for (int ci = 0; ci < 12; ++ci)
                acc += m_buf[w][ci] * Wu_lds[(12 + ci) * 12 + lane];
            out[(size_t)g * 12 + lane] = 1.0f / (1.0f + __expf(-acc));
        }
    }
}

// ---------------------------------------------------------------------------
extern "C" void kernel_launch(void* const* d_in, const int* in_sizes, int n_in,
                              void* d_out, int out_size, void* d_ws, size_t ws_size,
                              hipStream_t stream) {
    (void)in_sizes; (void)n_in; (void)out_size; (void)ws_size;

    const float* x    = (const float*)d_in[0];
    const float* d1   = (const float*)d_in[1];
    const float* d0   = (const float*)d_in[2];
    const float* dm1  = (const float*)d_in[3];
    // d_in[4] = mask (unused by reference math)
    const float* We   = (const float*)d_in[5];
    const float* be   = (const float*)d_in[6];
    const float* Wm1  = (const float*)d_in[7];
    const float* bm1  = (const float*)d_in[8];
    const float* Wu1  = (const float*)d_in[9];
    const float* bu1  = (const float*)d_in[10];
    const float* Wm0  = (const float*)d_in[11];
    const float* bm0  = (const float*)d_in[12];
    const float* Wu0  = (const float*)d_in[13];
    const float* bu0  = (const float*)d_in[14];
    const float* Wmm1 = (const float*)d_in[15];
    const float* bmm1 = (const float*)d_in[16];
    const float* Wum1 = (const float*)d_in[17];
    const float* bum1 = (const float*)d_in[18];

    float*    out = (float*)d_out;
    _Float16* h16 = (_Float16*)d_ws;   // 65536 * 16 halves = 2 MB

    embed_kernel<<<NTOT / 256, 256, 0, stream>>>(x, We, be, h16);

    branch_kernel<<<BR_BLOCKS, 256, 0, stream>>>(d1,  h16, Wm1,  bm1,  Wu1,  bu1,
                                                 out);
    branch_kernel<<<BR_BLOCKS, 256, 0, stream>>>(d0,  h16, Wm0,  bm0,  Wu0,  bu0,
                                                 out + (size_t)NTOT * 12);
    branch_kernel<<<BR_BLOCKS, 256, 0, stream>>>(dm1, h16, Wmm1, bmm1, Wum1, bum1,
                                                 out + (size_t)2 * NTOT * 12);
}
